// MultiHeadAttention_82145544503758
// MI455X (gfx1250) — matrix operand order repaired and checked by reading, not compiled, not run on hardware
//
#include <hip/hip_runtime.h>
#include <hip/hip_bf16.h>
#include <stdint.h>

// ---------------------------------------------------------------------------
// MHA for MI455X (gfx1250): bf16 WMMA GEMMs + flash-attention (no S x S buf),
// with Tensor Data Mover (TDM) tile staging (6-arg builtin, clang-23 lane).
//   B=2, S=2048, H=1024, NH=16, HD=64
// ---------------------------------------------------------------------------

typedef __bf16 bf16_t;
typedef __attribute__((ext_vector_type(16))) __bf16 v16bf;
typedef __attribute__((ext_vector_type(8)))  float  v8f;
typedef unsigned int u32x4 __attribute__((ext_vector_type(4)));
typedef int          i32x8 __attribute__((ext_vector_type(8)));
typedef int          i32x4 __attribute__((ext_vector_type(4)));

#define WMMA_BF16(a, b, c) \
  __builtin_amdgcn_wmma_f32_16x16x32_bf16(false, (a), false, (b), (short)0, (c), false, false)

#if defined(__has_builtin)
#if __has_builtin(__builtin_amdgcn_tensor_load_to_lds) && \
    __has_builtin(__builtin_amdgcn_s_wait_tensorcnt)
#define USE_TDM 1
#endif
#endif
#ifndef USE_TDM
#define USE_TDM 0
#endif

constexpr int kB  = 2;
constexpr int kS  = 2048;
constexpr int kH  = 1024;
constexpr int kNH = 16;
constexpr int kHD = 64;
constexpr int kH3 = 3 * kH;              // 3072
constexpr float kInvScale = 0.125f;      // 1/sqrt(64)

#if USE_TDM
// ---------------------------------------------------------------------------
// TDM 2D tile load: global (row-major, element=2B) -> LDS, with LDS row
// padding applied by the TDM pad engine (ISA 8.4: pad_interval/pad_amount).
//   tile_cols/tile_rows : tile shape in elements
//   row_stride          : tensor dim0 stride in elements
//   pad_int_code        : DWORDs-before-pad = 2^(code+1)
//   pad_amt_code        : pad DWORDs = code+1
// Groups 2/3 (and the trailing group) are zero: 2D tensor, no iterate/gather.
// Issued per-wave; EXEC ignored by TDM; tracked with TENSORcnt.
// ---------------------------------------------------------------------------
__device__ inline void tdm_load_2d(const void* gptr, unsigned lds_addr,
                                   unsigned tile_cols, unsigned tile_rows,
                                   unsigned row_stride, unsigned tensor_rows,
                                   unsigned pad_int_code, unsigned pad_amt_code) {
  unsigned long long ga = (unsigned long long)(uintptr_t)gptr;
  u32x4 g0;
  g0[0] = 1u;                                              // count=1, user mode
  g0[1] = lds_addr;                                        // LDS byte address
  g0[2] = (unsigned)ga;                                    // global_addr[31:0]
  g0[3] = (unsigned)((ga >> 32) & 0x01FFFFFFull)           // global_addr[56:32]
          | 0x80000000u;                                   // type=2 ("image")
  i32x8 g1;
  g1[0] = (int)((1u << 16)                                 // data_size: 2 bytes
                | (1u << 20)                               // pad_enable
                | (pad_int_code << 22) | (pad_amt_code << 25));
  g1[1] = (int)((row_stride & 0xFFFFu) << 16);             // tensor_dim0[15:0]
  g1[2] = (int)(((row_stride >> 16) & 0xFFFFu)             // tensor_dim0[31:16]
                | ((tensor_rows & 0xFFFFu) << 16));        // tensor_dim1[15:0]
  g1[3] = (int)(((tensor_rows >> 16) & 0xFFFFu)            // tensor_dim1[31:16]
                | (tile_cols << 16));                      // tile_dim0
  g1[4] = (int)tile_rows;                                  // tile_dim1 (tile_dim2=0)
  g1[5] = (int)row_stride;                                 // tensor_dim0_stride[31:0]
  g1[6] = 0;                                               // stride hi / dim1_stride
  g1[7] = 0;
  i32x4 z4 = {0, 0, 0, 0};
  i32x8 z8 = {0, 0, 0, 0, 0, 0, 0, 0};
  __builtin_amdgcn_tensor_load_to_lds(g0, g1, z4, z4, z8, 0);
}
#endif

__device__ inline unsigned lds_byte_addr(const void* p) {
  // __shared__ generic pointer = {SHARED_BASE, lds_offset}; low 32b = LDS addr
  return (unsigned)(unsigned long long)(uintptr_t)p;
}

// ---------------------------------------------------------------------------
// fp32 -> bf16 cast, 4 elements/thread
// ---------------------------------------------------------------------------
__global__ __launch_bounds__(256) void cast_f32_bf16(const float* __restrict__ in,
                                                     bf16_t* __restrict__ out, int n) {
  int i = (blockIdx.x * 256 + threadIdx.x) * 4;
  if (i + 3 < n) {
    float4 v = *(const float4*)(in + i);
    union { bf16_t b[4]; uint2 u; } o;
    o.b[0] = (bf16_t)v.x; o.b[1] = (bf16_t)v.y;
    o.b[2] = (bf16_t)v.z; o.b[3] = (bf16_t)v.w;
    *(uint2*)(out + i) = o.u;
  }
}

// ---------------------------------------------------------------------------
// C[M,N] = A[M,K] @ W[N,K]^T   (A,W bf16 row-major; OutT = bf16 or float)
// Block: 256 threads = 8 waves; tile BM=128 x BN=64, BK=32.
// Wave w owns rows [w*16, w*16+16) x all 64 cols: 4 WMMAs / k-step, A-frag
// reused across the 4 B-frags.
// ---------------------------------------------------------------------------
template <typename OutT>
__global__ __launch_bounds__(256) void gemm_xwt(const bf16_t* __restrict__ A,
                                                const bf16_t* __restrict__ W,
                                                OutT* __restrict__ C,
                                                int M, int N, int K) {
  constexpr int LDP = 40;                       // 64B data + 16B pad per row
  __shared__ bf16_t As[128][LDP];
  __shared__ bf16_t Ws[64][LDP];

  const int tid  = threadIdx.x;
  const int lane = tid & 31, wave = tid >> 5;
  const int nBlk = N / 64;
  const int mBase = (blockIdx.x / nBlk) * 128;
  const int nBase = (blockIdx.x % nBlk) * 64;
  const int frow = lane & 15, half = lane >> 4;

  v8f acc[4];
#pragma unroll
  for (int t = 0; t < 4; t++) acc[t] = (v8f){};

#if USE_TDM
  const unsigned ldsA = lds_byte_addr(&As[0][0]);
  const unsigned ldsW = lds_byte_addr(&Ws[0][0]);
#endif

  for (int k0 = 0; k0 < K; k0 += 32) {
    __syncthreads();
#if USE_TDM
    if (wave == 0) {
      // tile row = 32 elems = 64B = 16 DW -> pad_int=3; pad 16B = 4 DW -> pad_amt=3
      tdm_load_2d(A + (size_t)mBase * K + k0, ldsA, 32, 128, (unsigned)K, (unsigned)M, 3, 3);
      tdm_load_2d(W + (size_t)nBase * K + k0, ldsW, 32, 64,  (unsigned)K, (unsigned)N, 3, 3);
      __builtin_amdgcn_s_wait_tensorcnt(0);
    }
#else
    {
      const int i0 = tid * 2;
#pragma unroll
      for (int j = 0; j < 2; j++) {
        int i = i0 + j;
        int row = i >> 2, col = (i & 3) * 8;
        *(uint4*)&As[row][col] = *(const uint4*)(A + (size_t)(mBase + row) * K + k0 + col);
      }
      int wrow = tid >> 2, wcol = (tid & 3) * 8;
      *(uint4*)&Ws[wrow][wcol] = *(const uint4*)(W + (size_t)(nBase + wrow) * K + k0 + wcol);
    }
#endif
    __syncthreads();

    v16bf a;
    {
      uint4* ap = (uint4*)&a;                              // A-frag: 32B contiguous
      ap[0] = *(const uint4*)&As[wave * 16 + frow][half * 16];
      ap[1] = *(const uint4*)&As[wave * 16 + frow][half * 16 + 8];
    }
#pragma unroll
    for (int t = 0; t < 4; t++) {
      v16bf bfr;                                           // B-frag: 32B contiguous
      uint4* bp = (uint4*)&bfr;
      const uint4* w0 = (const uint4*)&Ws[t * 16 + frow][half * 16];
      bp[0] = w0[0]; bp[1] = w0[1];
      acc[t] = WMMA_BF16(a, bfr, acc[t]);
    }
  }

#pragma unroll
  for (int t = 0; t < 4; t++) {
#pragma unroll
    for (int r = 0; r < 8; r++) {
      int m = mBase + wave * 16 + r + 8 * half;
      int n = nBase + t * 16 + frow;
      C[(size_t)m * N + n] = (OutT)acc[t][r];
    }
  }
}

// ---------------------------------------------------------------------------
// Flash attention over bf16 QKV buffer [B*S, 3H] (q|k|v per token).
// Block: 256 thr = 8 waves; block owns 128 q-rows of one (b,h); wave owns 16.
// KV streamed in 32-row steps staged in LDS (K rows TDM/raw, V transposed).
// Online softmax in fp32; output written as bf16 [B*S, H].
// ---------------------------------------------------------------------------
__global__ __launch_bounds__(256) void flash_attn(const bf16_t* __restrict__ qkv,
                                                  const float* __restrict__ mask,
                                                  bf16_t* __restrict__ outb) {
  constexpr int nQB = kS / 128;                  // 16 q-blocks per (b,h)
  const int bh   = blockIdx.x / nQB;
  const int qblk = blockIdx.x % nQB;
  const int b = bh / kNH, h = bh % kNH;
  const int tid = threadIdx.x, lane = tid & 31, wave = tid >> 5;
  const int frow = lane & 15, half = lane >> 4;
  const int qt = qblk * 128 + wave * 16;

  __shared__ bf16_t Ks[32][72];                  // 128B data + 16B pad per row
  __shared__ bf16_t Vt[64][40];                  // V transposed, 80B stride
  __shared__ bf16_t Ps[8][16][40];               // per-wave P tile (C->A relayout)

  // Q fragments held in registers for the whole KV sweep (HD=64 -> 2 frags)
  v16bf qa0, qa1;
  {
    const bf16_t* qrow = qkv + (size_t)(b * kS + qt + frow) * kH3 + h * kHD;
    uint4* p0 = (uint4*)&qa0;
    p0[0] = *(const uint4*)(qrow + half * 16);
    p0[1] = *(const uint4*)(qrow + half * 16 + 8);
    uint4* p1 = (uint4*)&qa1;
    p1[0] = *(const uint4*)(qrow + 32 + half * 16);
    p1[1] = *(const uint4*)(qrow + 40 + half * 16);
  }

  float m_i[8], l_i[8];
#pragma unroll
  for (int r = 0; r < 8; r++) { m_i[r] = -1e30f; l_i[r] = 0.f; }
  v8f o0 = {}, o1 = {}, o2 = {}, o3 = {};

  const int lrow = tid >> 3, lcol = (tid & 7) * 8;   // 32 rows x 8 b128 loads
  const float* mbase = mask + (size_t)b * kS * kS;
#if USE_TDM
  const unsigned ldsK = lds_byte_addr(&Ks[0][0]);
#endif

  for (int j0 = 0; j0 < kS; j0 += 32) {
    __syncthreads();
#if USE_TDM
    if (wave == 0) {
      // tile row = 64 elems = 128B = 32 DW -> pad_int=4; pad 16B = 4 DW -> pad_amt=3
      tdm_load_2d(qkv + (size_t)(b * kS + j0) * kH3 + kH + h * kHD, ldsK,
                  64, 32, (unsigned)kH3, (unsigned)(kB * kS), 4, 3);
    }
#else
    *(uint4*)&Ks[lrow][lcol] =
        *(const uint4*)(qkv + (size_t)(b * kS + j0 + lrow) * kH3 + kH + h * kHD + lcol);
#endif
    // cooperative V tile load + transpose into Vt[hd][kv] (overlaps TDM)
    {
      union { uint4 u; bf16_t e[8]; } vv;
      vv.u = *(const uint4*)(qkv + (size_t)(b * kS + j0 + lrow) * kH3 + 2 * kH + h * kHD + lcol);
#pragma unroll
      for (int e = 0; e < 8; e++) Vt[lcol + e][lrow] = vv.e[e];
    }
#if USE_TDM
    if (wave == 0) __builtin_amdgcn_s_wait_tensorcnt(0);
#endif
    __syncthreads();

    // ---- scores: S16x32 = Q(16x64) @ K^T, 4 WMMAs ----
    v8f c0 = {}, c1 = {};
    {
      v16bf kb0, kb1, kb2, kb3;
      uint4* kp;
      const uint4* s;
      kp = (uint4*)&kb0; s = (const uint4*)&Ks[frow][half * 16];           kp[0] = s[0]; kp[1] = s[1];
      kp = (uint4*)&kb1; s = (const uint4*)&Ks[frow][32 + half * 16];      kp[0] = s[0]; kp[1] = s[1];
      kp = (uint4*)&kb2; s = (const uint4*)&Ks[16 + frow][half * 16];      kp[0] = s[0]; kp[1] = s[1];
      kp = (uint4*)&kb3; s = (const uint4*)&Ks[16 + frow][32 + half * 16]; kp[0] = s[0]; kp[1] = s[1];
      c0 = WMMA_BF16(qa0, kb0, c0);
      c0 = WMMA_BF16(qa1, kb1, c0);
      c1 = WMMA_BF16(qa0, kb2, c1);
      c1 = WMMA_BF16(qa1, kb3, c1);
    }

    // ---- online softmax (row M lives across 16 lanes of vgpr r) ----
#pragma unroll
    for (int r = 0; r < 8; r++) {
      int q = qt + r + 8 * half;
      const float* mp = mbase + (size_t)q * kS + j0;
      float s0 = c0[r] * kInvScale + mp[frow];
      float s1 = c1[r] * kInvScale + mp[16 + frow];
      float v = fmaxf(s0, s1);
      v = fmaxf(v, __shfl_xor(v, 1));
      v = fmaxf(v, __shfl_xor(v, 2));
      v = fmaxf(v, __shfl_xor(v, 4));
      v = fmaxf(v, __shfl_xor(v, 8));
      float mn = fmaxf(m_i[r], v);
      float alpha = __expf(m_i[r] - mn);
      float p0 = __expf(s0 - mn);
      float p1 = __expf(s1 - mn);
      float sum = p0 + p1;
      sum += __shfl_xor(sum, 1);
      sum += __shfl_xor(sum, 2);
      sum += __shfl_xor(sum, 4);
      sum += __shfl_xor(sum, 8);
      l_i[r] = l_i[r] * alpha + sum;
      m_i[r] = mn;
      o0[r] *= alpha; o1[r] *= alpha; o2[r] *= alpha; o3[r] *= alpha;
      int Mr = r + 8 * half;
      Ps[wave][Mr][frow]      = (bf16_t)p0;    // C-layout -> LDS (per-wave, no barrier)
      Ps[wave][Mr][16 + frow] = (bf16_t)p1;
    }

    // ---- P(16x32) @ V(32x64): 4 WMMAs into persistent accumulators ----
    v16bf pa;
    {
      uint4* pp = (uint4*)&pa;
      pp[0] = *(const uint4*)&Ps[wave][frow][half * 16];
      pp[1] = *(const uint4*)&Ps[wave][frow][half * 16 + 8];
    }
    {
      v16bf vb0, vb1, vb2, vb3;
      uint4* vp;
      const uint4* t;
      vp = (uint4*)&vb0; t = (const uint4*)&Vt[frow][half * 16];      vp[0] = t[0]; vp[1] = t[1];
      vp = (uint4*)&vb1; t = (const uint4*)&Vt[16 + frow][half * 16]; vp[0] = t[0]; vp[1] = t[1];
      vp = (uint4*)&vb2; t = (const uint4*)&Vt[32 + frow][half * 16]; vp[0] = t[0]; vp[1] = t[1];
      vp = (uint4*)&vb3; t = (const uint4*)&Vt[48 + frow][half * 16]; vp[0] = t[0]; vp[1] = t[1];
      o0 = WMMA_BF16(pa, vb0, o0);
      o1 = WMMA_BF16(pa, vb1, o1);
      o2 = WMMA_BF16(pa, vb2, o2);
      o3 = WMMA_BF16(pa, vb3, o3);
    }
  }

  // epilogue: normalize and emit bf16 attn output [B*S, H]
#pragma unroll
  for (int r = 0; r < 8; r++) {
    float inv = 1.0f / l_i[r];
    int tok = b * kS + qt + r + 8 * half;
    bf16_t* op = outb + (size_t)tok * kH + h * kHD + frow;
    op[0]  = (bf16_t)(o0[r] * inv);
    op[16] = (bf16_t)(o1[r] * inv);
    op[32] = (bf16_t)(o2[r] * inv);
    op[48] = (bf16_t)(o3[r] * inv);
  }
}

// ---------------------------------------------------------------------------
extern "C" void kernel_launch(void* const* d_in, const int* in_sizes, int n_in,
                              void* d_out, int out_size, void* d_ws, size_t ws_size,
                              hipStream_t stream) {
  const float* x    = (const float*)d_in[0];   // [B,S,H]
  const float* mask = (const float*)d_in[1];   // [B,S,S]
  const float* wqkv = (const float*)d_in[2];   // [3H,H]
  const float* wo   = (const float*)d_in[3];   // [H,H]
  float* out = (float*)d_out;                  // [B,S,H]

  char* ws = (char*)d_ws;
  size_t off = 0;
  bf16_t* Xb   = (bf16_t*)(ws + off); off += (size_t)kB * kS * kH * 2;    // 8 MB
  bf16_t* Wqb  = (bf16_t*)(ws + off); off += (size_t)kH3 * kH * 2;        // 6 MB
  bf16_t* Wob  = (bf16_t*)(ws + off); off += (size_t)kH * kH * 2;         // 2 MB
  bf16_t* QKVb = (bf16_t*)(ws + off); off += (size_t)kB * kS * kH3 * 2;   // 24 MB
  bf16_t* Ab   = (bf16_t*)(ws + off); off += (size_t)kB * kS * kH * 2;    // 8 MB

  const int M = kB * kS;                        // 4096
  const int nX = M * kH, nWq = kH3 * kH, nWo = kH * kH;

  cast_f32_bf16<<<nX / 4 / 256, 256, 0, stream>>>(x, Xb, nX);
  cast_f32_bf16<<<nWq / 4 / 256, 256, 0, stream>>>(wqkv, Wqb, nWq);
  cast_f32_bf16<<<nWo / 4 / 256, 256, 0, stream>>>(wo, Wob, nWo);

  // QKV projection: [4096,1024] @ [3072,1024]^T -> bf16 [4096,3072]
  gemm_xwt<bf16_t><<<(M / 128) * (kH3 / 64), 256, 0, stream>>>(Xb, Wqb, QKVb, M, kH3, kH);

  // flash attention -> bf16 [4096,1024]
  flash_attn<<<kB * kNH * (kS / 128), 256, 0, stream>>>(QKVb, mask, Ab);

  // output projection: [4096,1024] @ [1024,1024]^T -> fp32 d_out
  gemm_xwt<float><<<(M / 128) * (kH / 64), 256, 0, stream>>>(Ab, Wob, out, M, kH, kH);
}
